// LSTM_62079457296916
// MI455X (gfx1250) — compile-verified, matrix operand order checked by reading, not run on hardware
//
#include <hip/hip_runtime.h>
#include <hip/hip_bf16.h>

#define SEQ    32
#define BATCHN 32768
#define POSE   34
#define HDIM   16
#define PREDL  15
#define NWAVES 4
#define BTILE  (NWAVES * 16)

typedef __attribute__((ext_vector_type(16))) _Float16 v16h;
typedef __attribute__((ext_vector_type(8)))  _Float16 v8h;
typedef __attribute__((ext_vector_type(8)))  float    v8f;

// Fast activations: v_exp_f32 + v_rcp_f32 (no IEEE div expansion).
__device__ __forceinline__ float sig_(float x) {
  return __builtin_amdgcn_rcpf(1.f + __expf(-x));
}
// tanh via exp+rcp, saturates correctly at +/-1 for large |x|
__device__ __forceinline__ float th_(float x) {
  return 1.f - 2.f * __builtin_amdgcn_rcpf(__expf(2.f * x) + 1.f);
}
__device__ __forceinline__ v8f splat8(float v) { v8f r = {v, v, v, v, v, v, v, v}; return r; }

// Load a 16x32 f16 WMMA fragment (A or B) from an LDS tile stored row-major
// with `stride` halfs per row. Matches CDNA5 16-bit A/B layout:
//   lanes 0-15 : row = lane,    K = k0+{0..7}  (v0..3),  k0+{16..23} (v4..7)
//   lanes 16-31: row = lane-16, K = k0+{8..15} (v0..3),  k0+{24..31} (v4..7)
__device__ __forceinline__ v16h ldfrag(const _Float16* base, int stride, int lane, int k0) {
  const _Float16* p = base + (lane & 15) * stride + k0 + ((lane & 16) ? 8 : 0);
  v16h f;
  *(v8h*)&f       = *(const v8h*)p;        // ds_load_b128
  *((v8h*)&f + 1) = *(const v8h*)(p + 16); // ds_load_b128
  return f;
}

#define WMMA(A, B, C) __builtin_amdgcn_wmma_f32_16x16x32_f16(false, (A), false, (B), (short)0, (C), false, false)

// One fused LSTM cell step for a 16-batch tile:
//   gates[16x64] = Xcomb[16x64] @ Wcomb[64x64]^T(+bias), packed K = [x(34)|h(16)|0(14)]
// 8 WMMAs (4 gate N-tiles x 2 K-chunks), then elementwise i,f,g,o activations.
__device__ __forceinline__ void cell_step(const _Float16* X, const _Float16* W,
                                          const float* B, int lane, v8f& h, v8f& c) {
  v16h a0 = ldfrag(X, 64, lane, 0);
  v16h a1 = ldfrag(X, 64, lane, 32);
  v8f acc[4];
#pragma unroll
  for (int nt = 0; nt < 4; ++nt) {
    v8f acc_nt = splat8(B[nt * 16 + (lane & 15)]);
    v16h b0 = ldfrag(W + nt * 16 * 64, 64, lane, 0);
    v16h b1 = ldfrag(W + nt * 16 * 64, 64, lane, 32);
    acc_nt = WMMA(a0, b0, acc_nt);
    acc_nt = WMMA(a1, b1, acc_nt);
    acc[nt] = acc_nt;
  }
#pragma unroll
  for (int r = 0; r < 8; ++r) {
    float ig = sig_(acc[0][r]);
    float fg = sig_(acc[1][r]);
    float gg = th_(acc[2][r]);
    float og = sig_(acc[3][r]);
    float cn = fmaf(fg, c[r], ig * gg);
    c[r] = cn;
    h[r] = og * th_(cn);
  }
}

__global__ __launch_bounds__(128) void lstm_traj_kernel(
    const float* __restrict__ obs,
    const float* encWih, const float* encWhh, const float* encbih, const float* encbhh,
    const float* decWih, const float* decWhh, const float* decbih, const float* decbhh,
    const float* crWih,  const float* crWhh,  const float* crbih,  const float* crbhh,
    const float* fcW, const float* fcb, const float* fccW, const float* fccb,
    const float* mlpW, const float* mlpb, const float* embW, const float* embb,
    float* __restrict__ out) {
  // Combined weights [n=4H][k=64] f16: k<34 -> Wih, 34<=k<50 -> Whh, else 0
  __shared__ _Float16 sWenc[64 * 64];
  __shared__ _Float16 sWdec[64 * 64];
  __shared__ _Float16 sWcr [64 * 64];
  __shared__ _Float16 sWfc [48 * 32];   // fc_W padded N->48, K->32
  __shared__ _Float16 sWmlp[16 * 32];   // mlp_W padded K->32
  __shared__ float sBenc[64], sBdec[64], sBcr[64], sBfc[48], sBmlp[16];
  __shared__ float sFcc[34];            // [0..31] = fcc_W rows, [32..33] = fcc_b
  __shared__ float sEmb[102];           // [2n+j] = emb_W[n][j], [68+n] = emb_b[n]
  // Per-wave packed input rows [16 x 64] f16: [x(34)|h(16)|pad(14)]
  __shared__ _Float16 sXd[NWAVES][16 * 64];  // decoder/encoder stream
  __shared__ _Float16 sXc[NWAVES][16 * 64];  // crossing stream
  __shared__ _Float16 sHl[NWAVES][16 * 32];  // h with zero-padded K for mlp/fc

  const int tid = threadIdx.x;

  // ---- one-time weight pack (f32 -> f16, LDS) ----
  for (int i = tid; i < 64 * 64; i += 128) {
    int n = i >> 6, k = i & 63;
    float e = 0.f, d = 0.f, cw = 0.f;
    if (k < 34)      { int j = n * 34 + k;        e = encWih[j]; d = decWih[j]; cw = crWih[j]; }
    else if (k < 50) { int j = n * 16 + (k - 34); e = encWhh[j]; d = decWhh[j]; cw = crWhh[j]; }
    sWenc[i] = (_Float16)e; sWdec[i] = (_Float16)d; sWcr[i] = (_Float16)cw;
  }
  for (int i = tid; i < 48 * 32; i += 128) {
    int n = i >> 5, k = i & 31;
    sWfc[i] = (_Float16)((n < 34 && k < 16) ? fcW[n * 16 + k] : 0.f);
  }
  for (int i = tid; i < 16 * 32; i += 128) {
    int n = i >> 5, k = i & 31;
    sWmlp[i] = (_Float16)((k < 16) ? mlpW[n * 16 + k] : 0.f);
  }
  for (int i = tid; i < 64; i += 128) {
    sBenc[i] = encbih[i] + encbhh[i];
    sBdec[i] = decbih[i] + decbhh[i];
    sBcr[i]  = crbih[i]  + crbhh[i];
  }
  for (int i = tid; i < 48; i += 128) sBfc[i] = (i < 34) ? fcb[i] : 0.f;
  if (tid < 16) sBmlp[tid] = mlpb[tid];
  if (tid < 32) sFcc[tid] = fccW[tid];
  if (tid < 2)  sFcc[32 + tid] = fccb[tid];
  for (int i = tid; i < 68; i += 128) sEmb[i] = embW[i];
  for (int i = tid; i < 34; i += 128) sEmb[68 + i] = embb[i];

  const int wv = tid >> 5, lane = tid & 31;
  const int nl = lane & 15;
  const int hib = (lane & 16) ? 8 : 0;  // C/D row offset for upper half-lanes
  _Float16* Xd = sXd[wv];
  _Float16* Xc = sXc[wv];
  _Float16* Hl = sHl[wv];

  // zero pad regions (cols 50..63 of X rows; cols 16..31 of Hl rows)
  for (int i = lane; i < 16 * 14; i += 32) {
    int row = i / 14, col = 50 + (i % 14);
    Xd[row * 64 + col] = (_Float16)0.f;
    Xc[row * 64 + col] = (_Float16)0.f;
  }
  for (int i = lane; i < 16 * 16; i += 32) {
    int row = i >> 4, col = 16 + (i & 15);
    Hl[row * 32 + col] = (_Float16)0.f;
  }
  __syncthreads();

  const long B0 = (long)blockIdx.x * BTILE + wv * 16;

  // ---- encoder scan ----
  v8f h = splat8(0.f), c = splat8(0.f);
  for (int t = 0; t < SEQ; ++t) {
#pragma unroll
    for (int r = 0; r < 8; ++r) Xd[(r + hib) * 64 + 34 + nl] = (_Float16)h[r];
    const float* xp = obs + ((long)t * BATCHN + B0) * POSE;
    for (int i = lane; i < 16 * POSE; i += 32) {
      int row = i / POSE, col = i - row * POSE;
      Xd[row * 64 + col] = (_Float16)xp[i];
    }
    if (t + 1 < SEQ) __builtin_prefetch(xp + (long)BATCHN * POSE, 0, 1);
    __syncthreads();
    cell_step(Xd, sWenc, sBenc, lane, h, c);
    __syncthreads();
  }

  // final h -> Xd (dec cell input uses [obs[-1], h]) and Hl (mlp input)
#pragma unroll
  for (int r = 0; r < 8; ++r) {
    Xd[(r + hib) * 64 + 34 + nl] = (_Float16)h[r];
    Hl[(r + hib) * 32 + nl]      = (_Float16)h[r];
  }
  __syncthreads();

  // hc = h @ mlp_W^T + mlp_b : one WMMA with zero-padded K
  v8f hc, cc = splat8(0.f);
  {
    v16h am = ldfrag(Hl, 32, lane, 0);
    v16h bm = ldfrag(sWmlp, 32, lane, 0);
    v8f a = splat8(sBmlp[nl]);
    hc = WMMA(am, bm, a);
  }
  // init crossing stream: last_cr = obs[-1] (already f16 in Xd), plus hc
  for (int i = lane; i < 16 * POSE; i += 32) {
    int row = i / POSE, col = i - row * POSE;
    Xc[row * 64 + col] = Xd[row * 64 + col];
  }
#pragma unroll
  for (int r = 0; r < 8; ++r) Xc[(r + hib) * 64 + 34 + nl] = (_Float16)hc[r];
  __syncthreads();

  float* predS = out;
  float* predC = out + (long)PREDL * BATCHN * POSE;

  // ---- decoder scan ----
  for (int t = 0; t < PREDL; ++t) {
    // crossing LSTM cell
    cell_step(Xc, sWcr, sBcr, lane, hc, cc);
#pragma unroll
    for (int r = 0; r < 8; ++r) Xc[(r + hib) * 64 + 34 + nl] = (_Float16)hc[r];
    __syncthreads();

    // fcc + softmax + emb: one batch row per lane (lanes 0..15)
    if (lane < 16) {
      int row = lane;
      float z0 = sFcc[32], z1 = sFcc[33];
#pragma unroll
      for (int k = 0; k < 16; ++k) {
        float hv = (float)Xc[row * 64 + 34 + k];
        z0 = fmaf(hv, sFcc[k], z0);
        z1 = fmaf(hv, sFcc[16 + k], z1);
      }
      float m = fmaxf(z0, z1);
      float e0 = __expf(z0 - m), e1 = __expf(z1 - m);
      float inv = __builtin_amdgcn_rcpf(e0 + e1);
      float p0 = e0 * inv, p1 = e1 * inv;
      long ob = ((long)t * BATCHN + B0 + row) * 2;
      predC[ob] = p0;
      predC[ob + 1] = p1;
      for (int n = 0; n < POSE; ++n) {  // next_cr -> Xc input cols for next step
        float v = fmaf(p0, sEmb[2 * n], fmaf(p1, sEmb[2 * n + 1], sEmb[68 + n]));
        Xc[row * 64 + n] = (_Float16)v;
      }
    }

    // decoder LSTM cell
    cell_step(Xd, sWdec, sBdec, lane, h, c);
#pragma unroll
    for (int r = 0; r < 8; ++r) {
      Xd[(r + hib) * 64 + 34 + nl] = (_Float16)h[r];
      Hl[(r + hib) * 32 + nl]      = (_Float16)h[r];
    }
    __syncthreads();

    // curr_s = h @ fc_W^T + fc_b : 3 padded WMMA N-tiles; write out + feed back
    v16h ah = ldfrag(Hl, 32, lane, 0);
#pragma unroll
    for (int nt = 0; nt < 3; ++nt) {
      v8f a = splat8(sBfc[nt * 16 + nl]);
      v16h bf = ldfrag(sWfc + nt * 16 * 32, 32, lane, 0);
      v8f s = WMMA(ah, bf, a);
      int n = nt * 16 + nl;
      if (n < POSE) {
#pragma unroll
        for (int r = 0; r < 8; ++r) {
          int M = r + hib;
          float v = s[r];
          predS[((long)t * BATCHN + B0 + M) * POSE + n] = v;
          Xd[M * 64 + n] = (_Float16)v;  // last_s for next step
        }
      }
    }
    __syncthreads();
  }
}

extern "C" void kernel_launch(void* const* d_in, const int* in_sizes, int n_in,
                              void* d_out, int out_size, void* d_ws, size_t ws_size,
                              hipStream_t stream) {
  (void)in_sizes; (void)n_in; (void)out_size; (void)d_ws; (void)ws_size;
  const float* obs    = (const float*)d_in[0];
  const float* encWih = (const float*)d_in[1];
  const float* encWhh = (const float*)d_in[2];
  const float* encbih = (const float*)d_in[3];
  const float* encbhh = (const float*)d_in[4];
  const float* decWih = (const float*)d_in[5];
  const float* decWhh = (const float*)d_in[6];
  const float* decbih = (const float*)d_in[7];
  const float* decbhh = (const float*)d_in[8];
  const float* crWih  = (const float*)d_in[9];
  const float* crWhh  = (const float*)d_in[10];
  const float* crbih  = (const float*)d_in[11];
  const float* crbhh  = (const float*)d_in[12];
  const float* fcW    = (const float*)d_in[13];
  const float* fcb    = (const float*)d_in[14];
  const float* fccW   = (const float*)d_in[15];
  const float* fccb   = (const float*)d_in[16];
  const float* mlpW   = (const float*)d_in[17];
  const float* mlpb   = (const float*)d_in[18];
  const float* embW   = (const float*)d_in[19];
  const float* embb   = (const float*)d_in[20];
  // d_in[21] = pred_len (compile-time constant PREDL here)

  dim3 grid(BATCHN / BTILE);  // 512 blocks, each 4 waves x 16-batch tiles
  dim3 block(128);
  lstm_traj_kernel<<<grid, block, 0, stream>>>(
      obs, encWih, encWhh, encbih, encbhh,
      decWih, decWhh, decbih, decbhh,
      crWih, crWhh, crbih, crbhh,
      fcW, fcb, fccW, fccb, mlpW, mlpb, embW, embb,
      (float*)d_out);
}
